// RNN_44555990729392
// MI455X (gfx1250) — matrix operand order repaired and checked by reading, not compiled, not run on hardware
//
#include <hip/hip_runtime.h>
#include <hip/hip_bf16.h>
#include <stdint.h>
#include <stddef.h>

// Problem sizes (fixed by the reference).
#define BB 32
#define SS 1024
#define DD 512
#define HH 512
#define NG 1536               // 3 gates * H, concatenated [z | r | o]
#define MM (BB * SS)          // 32768 rows for the x-projection GEMM
#define KK 512                // contraction dim everywhere (D == H == 512)

#define REC_BLOCKS 24         // 24 WGs x 8 waves = 192 waves in the recurrent phase

typedef __attribute__((ext_vector_type(8)))  __bf16 v8bf;
typedef __attribute__((ext_vector_type(16))) __bf16 v16bf;
typedef __attribute__((ext_vector_type(8)))  float  v8f;

// ---------------- small device helpers ----------------

__device__ __forceinline__ unsigned short f2bf(float f) {
  // round-to-nearest-even fp32 -> bf16 (bit pattern)
  unsigned u = __float_as_uint(f);
  u += 0x7fffu + ((u >> 16) & 1u);
  return (unsigned short)(u >> 16);
}

// A-fragment (16x32 bf16, row-major source, ld = KK):
// lanes 0-15 (M=lane): VGPR0..3 = K[k0..k0+7], VGPR4..7 = K[k0+16..k0+23]
// lanes 16-31 (M=lane-16): K[k0+8..k0+15] and K[k0+24..k0+31]
__device__ __forceinline__ v16bf load_a_frag(const unsigned short* arow, int k0, int hi) {
  v8bf lo = *(const v8bf*)(arow + k0 + hi * 8);
  v8bf hh = *(const v8bf*)(arow + k0 + 16 + hi * 8);
  return __builtin_shufflevector(lo, hh, 0, 1, 2, 3, 4, 5, 6, 7,
                                 8, 9, 10, 11, 12, 13, 14, 15);
}

// B-fragment (32x16 bf16). W stored transposed as WT[N rows, K cols] so a
// lane's column of B is a row chunk: lane l holds N=n0+l,
// lanes 0-15 -> K[k0..k0+7],K[k0+16..k0+23]; lanes 16-31 -> K[k0+8..k0+15],K[k0+24..k0+31].
__device__ __forceinline__ v16bf load_b_frag(const unsigned short* WT, int nrow, int k0, int hi) {
  const unsigned short* brow = WT + (size_t)nrow * KK;
  v8bf lo = *(const v8bf*)(brow + k0 + hi * 8);
  v8bf hh = *(const v8bf*)(brow + k0 + 16 + hi * 8);
  return __builtin_shufflevector(lo, hh, 0, 1, 2, 3, 4, 5, 6, 7,
                                 8, 9, 10, 11, 12, 13, 14, 15);
}

__device__ __forceinline__ v8f wmma_bf16(v16bf a, v16bf b, v8f c) {
  return __builtin_amdgcn_wmma_f32_16x16x32_bf16(false, a, false, b, (short)0, c,
                                                 false, false);
}

__device__ __forceinline__ float fast_sigmoid(float x) {
  return 1.0f / (1.0f + __expf(-x));
}

// Device-wide barrier: monotonically increasing counter, release on arrival,
// acquire while spinning. Counter is zeroed by init_state before each
// persistent launch -> deterministic across graph replays.
__device__ __forceinline__ void grid_barrier(unsigned* bar, unsigned& phase, unsigned nwg) {
  __syncthreads();
  ++phase;
  if (threadIdx.x == 0) {
    __hip_atomic_fetch_add(bar, 1u, __ATOMIC_RELEASE, __HIP_MEMORY_SCOPE_AGENT);
    const unsigned target = phase * nwg;
    while (__hip_atomic_load(bar, __ATOMIC_ACQUIRE, __HIP_MEMORY_SCOPE_AGENT) < target) {
      __builtin_amdgcn_s_sleep(1);
    }
  }
  __syncthreads();
}

// ---------------- preparation kernels ----------------

// Build bf16 transposed weights for one layer:
//   WTx[g*H+n, k] = W_g[k, n]          (x part,  k in [0,512))
//   WTh[g*H+n, k] = W_g[512+k, n]      (h part)
// and the concatenated bias biascat[3H].
__global__ void prep_weights(const float* __restrict__ Wz, const float* __restrict__ Wr,
                             const float* __restrict__ Wo,
                             const float* __restrict__ bz, const float* __restrict__ br,
                             const float* __restrict__ bo,
                             unsigned short* __restrict__ WTx,
                             unsigned short* __restrict__ WTh,
                             float* __restrict__ biascat) {
  const int idx = blockIdx.x * blockDim.x + threadIdx.x;
  if (idx < NG) {
    const int g = idx / HH, n = idx % HH;
    const float* bsrc = (g == 0) ? bz : (g == 1) ? br : bo;
    biascat[idx] = bsrc[n];
  }
  if (idx >= NG * KK) return;
  const int ng = idx / KK;
  const int k  = idx % KK;
  const int g = ng / HH, n = ng % HH;
  const float* W = (g == 0) ? Wz : (g == 1) ? Wr : Wo;   // [(512+512), 512] row-major
  WTx[(size_t)ng * KK + k] = f2bf(W[(size_t)k * HH + n]);
  WTh[(size_t)ng * KK + k] = f2bf(W[(size_t)(512 + k) * HH + n]);
}

__global__ void cvt_bf16(const float* __restrict__ src, unsigned short* __restrict__ dst, int n) {
  const int stride = gridDim.x * blockDim.x;
  for (int i = blockIdx.x * blockDim.x + threadIdx.x; i < n; i += stride)
    dst[i] = f2bf(src[i]);
}

__global__ void init_state(const float* __restrict__ h0, int layer,
                           float* __restrict__ hbuf, unsigned short* __restrict__ hbf,
                           unsigned* __restrict__ bar) {
  const int i = blockIdx.x * blockDim.x + threadIdx.x;
  if (i == 0) *bar = 0u;                         // reset barrier every launch
  if (i < BB * HH) {
    const int b = i >> 9, j = i & 511;
    const float v = h0[(size_t)(b * 2 + layer) * HH + j];
    hbuf[i] = v;
    hbf[i]  = f2bf(v);
  }
}

// ---------------- x-projection GEMM ----------------
// Apre[m, n] = sum_k Xbf[m, k] * WT[n, k] + biascat[n],  m in [0,32768), n in [0,1536)
// One wave per 16(M) x 64(N) tile: A-fragment reused across 4 B-fragments.
__global__ void __launch_bounds__(256) xproj_gemm(const unsigned short* __restrict__ Xbf,
                                                  const unsigned short* __restrict__ WT,
                                                  const float* __restrict__ biascat,
                                                  float* __restrict__ Apre) {
  const int lane = threadIdx.x & 31;
  const int l = lane & 15, hi = lane >> 4;
  const int gw = blockIdx.x * 8 + (threadIdx.x >> 5);
  const int mt = gw / 24, ns = gw % 24;          // (32768/16) x (1536/64) tiles
  const int m0 = mt * 16, n0 = ns * 64;

  v8f c[4] = {v8f{}, v8f{}, v8f{}, v8f{}};
  const unsigned short* arow = Xbf + (size_t)(m0 + l) * KK;

#pragma unroll 2
  for (int k0 = 0; k0 < KK; k0 += 32) {
    const v16bf a = load_a_frag(arow, k0, hi);
#pragma unroll
    for (int j = 0; j < 4; ++j) {
      const v16bf b = load_b_frag(WT, n0 + j * 16 + l, k0, hi);
      c[j] = wmma_bf16(a, b, c[j]);
    }
  }

#pragma unroll
  for (int j = 0; j < 4; ++j) {
    const int n = n0 + j * 16 + l;
    const float bias = biascat[n];
    float* orow = Apre + (size_t)(m0 + hi * 8) * NG + n;
#pragma unroll
    for (int r = 0; r < 8; ++r)
      orow[(size_t)r * NG] = c[j][r] + bias;     // C/D layout: VGPR r -> M = r + hi*8
  }
}

// ---------------- persistent recurrent kernel ----------------
// Grid: 24 blocks x 256 threads = 192 waves. Each wave owns ONE 16x16 output
// tile and pins its entire recurrent-weight panel (16 cols x 512 K of bf16 =
// 16 KB = 128 VGPRs) in registers for the whole time loop -> zero weight
// memory traffic on the sequential critical path.
//   waves   0..127 : z|r gates (2 M-tiles x 64 N-strips of 16)
//   waves 128..191 : candidate gate (2 M-tiles x 32 N-strips of 16)
__global__ void __launch_bounds__(256) gru_recurrent(
    const float* __restrict__ Apre, const unsigned short* __restrict__ WTh,
    float* __restrict__ hbuf, unsigned short* __restrict__ hbf,
    float* __restrict__ zbuf, unsigned short* __restrict__ rhbf,
    unsigned* __restrict__ bar,
    float* __restrict__ y_f32, unsigned short* __restrict__ y_bf,
    float* __restrict__ hlast) {
  const int lane = threadIdx.x & 31;
  const int l = lane & 15, hi = lane >> 4;
  const int w = blockIdx.x * 8 + (threadIdx.x >> 5);   // 0..191
  const unsigned nwg = gridDim.x;
  unsigned phase = 0;

  const bool s1 = (w < 128);                     // uniform per wave
  // stage-1 tile
  const int mt1 = (w >> 6) & 1;
  const int n0_1 = (w & 63) * 16;                // column in [0, 1024)
  const bool is_z = (n0_1 < HH);
  // stage-2 tile
  const int v2 = w - 128;
  const int mt2 = (v2 >> 5) & 1;
  const int n0_2 = (v2 & 31) * 16;               // column within the o gate

  // Preload this wave's weight panel into registers (role-dependent row).
  const int brow = (s1 ? n0_1 : 1024 + n0_2) + l;
  v16bf bw[16];
#pragma unroll
  for (int kk = 0; kk < 16; ++kk)
    bw[kk] = load_b_frag(WTh, brow, kk * 32, hi);

  for (int t = 0; t < SS; ++t) {
    // -------- stage 1: update/reset gates --------
    if (s1) {
      v8f c = {};
      const unsigned short* arow = hbf + (size_t)(mt1 * 16 + l) * KK;
#pragma unroll
      for (int kk = 0; kk < 16; ++kk)
        c = wmma_bf16(load_a_frag(arow, kk * 32, hi), bw[kk], c);

      const int n = n0_1 + l;
#pragma unroll
      for (int r = 0; r < 8; ++r) {
        const int bidx = mt1 * 16 + r + hi * 8;  // batch row
        const float pre = c[r] + Apre[((size_t)bidx * SS + t) * NG + n];
        const float g = fast_sigmoid(pre);
        if (is_z) {
          zbuf[bidx * HH + n] = g;
        } else {
          const int col = n - HH;
          rhbf[bidx * HH + col] = f2bf(g * hbuf[bidx * HH + col]);
        }
        if (t + 1 < SS)                          // hide HBM latency of streamed Apre
          __builtin_prefetch(&Apre[((size_t)bidx * SS + t + 1) * NG + n], 0, 0);
      }
    }
    grid_barrier(bar, phase, nwg);

    // -------- stage 2: candidate + state update --------
    if (!s1) {
      v8f c = {};
      const unsigned short* arow = rhbf + (size_t)(mt2 * 16 + l) * KK;
#pragma unroll
      for (int kk = 0; kk < 16; ++kk)
        c = wmma_bf16(load_a_frag(arow, kk * 32, hi), bw[kk], c);

      const int n = n0_2 + l;                    // column within H
#pragma unroll
      for (int r = 0; r < 8; ++r) {
        const int bidx = mt2 * 16 + r + hi * 8;
        const float pre = c[r] + Apre[((size_t)bidx * SS + t) * NG + 1024 + n];
        const float ht = tanhf(pre);
        const float z = zbuf[bidx * HH + n];
        const float hprev = hbuf[bidx * HH + n];
        const float hn = (1.0f - z) * ht + z * hprev;
        hbuf[bidx * HH + n] = hn;
        hbf[bidx * HH + n]  = f2bf(hn);
        if (y_f32) y_f32[((size_t)bidx * SS + t) * HH + n] = hn;
        if (y_bf)  y_bf [((size_t)bidx * SS + t) * HH + n] = f2bf(hn);
        if (t == SS - 1) hlast[bidx * (2 * HH) + n] = hn;  // [B, L, H] slice
        if (t + 1 < SS)
          __builtin_prefetch(&Apre[((size_t)bidx * SS + t + 1) * NG + 1024 + n], 0, 0);
      }
    }
    grid_barrier(bar, phase, nwg);
  }
}

// ---------------- host-side launch ----------------

static inline size_t align256(size_t v) { return (v + 255) & ~(size_t)255; }

extern "C" void kernel_launch(void* const* d_in, const int* in_sizes, int n_in,
                              void* d_out, int out_size, void* d_ws, size_t ws_size,
                              hipStream_t stream) {
  (void)in_sizes; (void)n_in; (void)out_size; (void)ws_size;
  const float* x   = (const float*)d_in[0];
  const float* h0  = (const float*)d_in[1];
  const float* Wz0 = (const float*)d_in[2];
  const float* bz0 = (const float*)d_in[3];
  const float* Wr0 = (const float*)d_in[4];
  const float* br0 = (const float*)d_in[5];
  const float* Wo0 = (const float*)d_in[6];
  const float* bo0 = (const float*)d_in[7];
  const float* Wz1 = (const float*)d_in[8];
  const float* bz1 = (const float*)d_in[9];
  const float* Wr1 = (const float*)d_in[10];
  const float* br1 = (const float*)d_in[11];
  const float* Wo1 = (const float*)d_in[12];
  const float* bo1 = (const float*)d_in[13];

  // Workspace carve (~241 MB).
  char* p = (char*)d_ws;
  float* Apre = (float*)p;            p += align256((size_t)MM * NG * sizeof(float));
  unsigned short* actbf = (unsigned short*)p;  // x in bf16, later overwritten by y0 bf16
                                      p += align256((size_t)MM * KK * sizeof(unsigned short));
  unsigned short* WTx0 = (unsigned short*)p;  p += align256((size_t)NG * KK * 2);
  unsigned short* WTh0 = (unsigned short*)p;  p += align256((size_t)NG * KK * 2);
  unsigned short* WTx1 = (unsigned short*)p;  p += align256((size_t)NG * KK * 2);
  unsigned short* WTh1 = (unsigned short*)p;  p += align256((size_t)NG * KK * 2);
  float* bias0 = (float*)p;           p += align256((size_t)NG * sizeof(float));
  float* bias1 = (float*)p;           p += align256((size_t)NG * sizeof(float));
  float* hbuf  = (float*)p;           p += align256((size_t)BB * HH * sizeof(float));
  float* zbuf  = (float*)p;           p += align256((size_t)BB * HH * sizeof(float));
  unsigned short* hbf  = (unsigned short*)p;  p += align256((size_t)BB * HH * 2);
  unsigned short* rhbf = (unsigned short*)p;  p += align256((size_t)BB * HH * 2);
  unsigned* bar = (unsigned*)p;       p += 256;

  float* y1_out    = (float*)d_out;                       // [B, S, H]
  float* hlast0    = (float*)d_out + (size_t)MM * HH;     // [B, L, H], layer-0 slice
  float* hlast1    = hlast0 + HH;                         // layer-1 slice

  const int prep_blocks = (NG * KK + 255) / 256;
  prep_weights<<<prep_blocks, 256, 0, stream>>>(Wz0, Wr0, Wo0, bz0, br0, bo0,
                                                WTx0, WTh0, bias0);
  prep_weights<<<prep_blocks, 256, 0, stream>>>(Wz1, Wr1, Wo1, bz1, br1, bo1,
                                                WTx1, WTh1, bias1);
  cvt_bf16<<<4096, 256, 0, stream>>>(x, actbf, MM * DD);

  const int gemm_blocks = (MM / 16) * (NG / 64) / 8;      // 6144 blocks, 8 waves each

  // ---- layer 0 ----
  xproj_gemm<<<gemm_blocks, 256, 0, stream>>>(actbf, WTx0, bias0, Apre);
  init_state<<<(BB * HH + 255) / 256, 256, 0, stream>>>(h0, 0, hbuf, hbf, bar);
  gru_recurrent<<<REC_BLOCKS, 256, 0, stream>>>(Apre, WTh0, hbuf, hbf, zbuf, rhbf, bar,
                                                nullptr, actbf /* y0 in bf16 */, hlast0);

  // ---- layer 1 (consumes y0 bf16 from actbf) ----
  xproj_gemm<<<gemm_blocks, 256, 0, stream>>>(actbf, WTx1, bias1, Apre);
  init_state<<<(BB * HH + 255) / 256, 256, 0, stream>>>(h0, 1, hbuf, hbf, bar);
  gru_recurrent<<<REC_BLOCKS, 256, 0, stream>>>(Apre, WTh1, hbuf, hbf, zbuf, rhbf, bar,
                                                y1_out, nullptr, hlast1);
}
